// Experts_19971597927215
// MI455X (gfx1250) — compile-verified, matrix operand order checked by reading, not run on hardware
//
#include <hip/hip_runtime.h>
#include <hip/hip_bf16.h>
#include <math.h>

// ---------------------------------------------------------------------------
// Grouped FFN, 8 experts, equal split.  All matmul work in bf16 WMMA.
// Pipeline:
//   pass0: X -> Xbf (bf16),  W1 -> W1T (bf16, [E,F,D]),  W2 -> W2T (bf16, [E,D,F])
//   pass1: H = gelu(Xbf @ W1 + b1)  (bf16, ws)        v_wmma_f32_16x16x32_bf16
//   pass2: OUT = H @ W2 + b2        (fp32)            v_wmma_f32_16x16x32_bf16
// Pre-converting removes the fp32->bf16 VALU stream from the GEMM K-loop
// (it was 16-64x redundant across blocks and dominated WMMA cycles), and the
// pre-transposed weights make every tile a pure 16B bf16 copy -> async
// global->LDS (GLOBAL_LOAD_ASYNC_TO_LDS_B128, ASYNCcnt) with double buffering.
// L2 (192MB) absorbs panel reuse; HBM traffic stays near-compulsory.
//
// Workspace layout (872,415,232 B total):
//   Xbf  [T,D]   bf16   67,108,864 B
//   W1T  [E,F,D] bf16  268,435,456 B
//   W2T  [E,D,F] bf16  268,435,456 B
//   H    [T,F]   bf16  268,435,456 B
// ---------------------------------------------------------------------------

typedef __attribute__((ext_vector_type(16))) __bf16 v16bf;
typedef __attribute__((ext_vector_type(8)))  __bf16 bf16x8;
typedef __attribute__((ext_vector_type(4)))  __bf16 bf16x4;
typedef __attribute__((ext_vector_type(2)))  __bf16 bf16x2;
typedef __attribute__((ext_vector_type(8)))  float  v8f;
typedef __attribute__((ext_vector_type(4)))  int    i32x4;

#define T_TOK  16384
#define NEXP   8
#define DMODEL 2048
#define DFF    8192
#define TPE    (T_TOK / NEXP)   // 2048

#define BM 128
#define BN 128
#define BK 32
#define LDSROW 40               // BK + 8 pad (80 B rows, 16B-aligned chunks)

// ---------------- scalar/packed fp32 -> bf16 ----------------
__device__ __forceinline__ __bf16 f2bf(float f) {
    unsigned u = __builtin_bit_cast(unsigned, f);
    unsigned r = u + 0x7FFFu + ((u >> 16) & 1u);
    return __builtin_bit_cast(__bf16, (unsigned short)(r >> 16));
}

__device__ __forceinline__ bf16x2 cvt2(float x, float y) {
#if __has_builtin(__builtin_amdgcn_cvt_pk_bf16_f32)
    return __builtin_amdgcn_cvt_pk_bf16_f32(x, y);
#else
    bf16x2 r; r[0] = f2bf(x); r[1] = f2bf(y); return r;
#endif
}

// gfx1250 has V_TANH_F32 (TRANS unit) -- use it when clang exposes a builtin.
__device__ __forceinline__ float fast_tanh(float x) {
#if __has_builtin(__builtin_amdgcn_tanhf)
    return __builtin_amdgcn_tanhf(x);
#elif __has_builtin(__builtin_amdgcn_tanh_f32)
    return __builtin_amdgcn_tanh_f32(x);
#else
    return tanhf(x);
#endif
}

__device__ __forceinline__ float gelu_tanh(float x) {
    float x3 = x * x * x;
    return 0.5f * x * (1.0f + fast_tanh(0.7978845608028654f * (x + 0.044715f * x3)));
}

// ---------------- async global->LDS 16B copy (guarded) ----------------
#if __has_builtin(__builtin_amdgcn_global_load_async_to_lds_b128) && \
    __has_builtin(__builtin_amdgcn_s_wait_asynccnt)
#define USE_ASYNC_LDS 1
#endif

__device__ __forceinline__ void copy16_g2l(const __bf16* g, __bf16* l) {
#ifdef USE_ASYNC_LDS
    // builtin signature: (i32x4 addrspace(1)*, i32x4 addrspace(3)*, imm offset, imm cpol)
    typedef __attribute__((address_space(1))) i32x4* gvp;
    typedef __attribute__((address_space(3))) i32x4* lvp;
    __builtin_amdgcn_global_load_async_to_lds_b128((gvp)g, (lvp)l, 0, 0);
#else
    *(bf16x8*)l = *(const bf16x8*)g;
#endif
}

__device__ __forceinline__ void wait_async_lds() {
#ifdef USE_ASYNC_LDS
    __builtin_amdgcn_s_wait_asynccnt(0);
#endif
}

// Stage a 128-row x 32-bf16 tile (8KB) into padded LDS: 512 x 16B chunks,
// 256 threads x 2 chunks.
__device__ __forceinline__ void stage_tile(const __bf16* __restrict__ g, size_t gstride,
                                           __bf16* l, int tid) {
#pragma unroll
    for (int p = 0; p < 2; ++p) {
        int chunk = tid + p * 256;       // 0..511
        int r     = chunk >> 2;          // 0..127
        int c8    = (chunk & 3) * 8;     // 0,8,16,24
        copy16_g2l(g + (size_t)r * gstride + c8, l + r * LDSROW + c8);
    }
}

// WMMA 16-bit A/B fragment: lane holds k = 8*(lane/16)+{0..7} (VGPRs 0-3)
// and k = 16+8*(lane/16)+{0..7} (VGPRs 4-7); row/col = lane%16.
__device__ __forceinline__ v16bf load_frag(const __bf16* rowbase, int kc0) {
    bf16x8 lo = *(const bf16x8*)(rowbase + kc0);
    bf16x8 hi = *(const bf16x8*)(rowbase + kc0 + 16);
    v16bf r;
#pragma unroll
    for (int i = 0; i < 8; ++i) { r[i] = lo[i]; r[8 + i] = hi[i]; }
    return r;
}

// ---------------------------------------------------------------------------
// pass 0a: elementwise fp32 -> bf16 (4 elems/thread)
// ---------------------------------------------------------------------------
__global__ __launch_bounds__(256)
void cvt_f32_bf16(const float* __restrict__ src, __bf16* __restrict__ dst) {
    size_t i = ((size_t)blockIdx.x * blockDim.x + threadIdx.x) * 4;
    float4 v = *(const float4*)(src + i);
    bf16x2 lo = cvt2(v.x, v.y), hi = cvt2(v.z, v.w);
    bf16x4 o; o[0] = lo[0]; o[1] = lo[1]; o[2] = hi[0]; o[3] = hi[1];
    *(bf16x4*)(dst + i) = o;
}

// ---------------------------------------------------------------------------
// pass 0b: per-expert transpose + convert: src [R,C] fp32 -> dst [C,R] bf16
// 64x64 tile via LDS.  grid: (C/64, R/64, E)
// ---------------------------------------------------------------------------
__global__ __launch_bounds__(256)
void transpose_cvt(const float* __restrict__ src, __bf16* __restrict__ dst,
                   int R, int C) {
    __shared__ __bf16 tile[64][72];
    const float* s = src + (size_t)blockIdx.z * R * C;
    __bf16*      d = dst + (size_t)blockIdx.z * R * C;
    const int r0 = blockIdx.y * 64, c0 = blockIdx.x * 64;
    const int t = threadIdx.x;
    const int tr = t >> 4, tc4 = (t & 15) * 4;
#pragma unroll
    for (int p = 0; p < 4; ++p) {
        int r = tr + p * 16;
        float4 v = *(const float4*)(s + (size_t)(r0 + r) * C + c0 + tc4);
        bf16x2 lo = cvt2(v.x, v.y), hi = cvt2(v.z, v.w);
        bf16x4 q; q[0] = lo[0]; q[1] = lo[1]; q[2] = hi[0]; q[3] = hi[1];
        *(bf16x4*)(&tile[r][tc4]) = q;
    }
    __syncthreads();
#pragma unroll
    for (int p = 0; p < 4; ++p) {
        int c = tr + p * 16;             // output row = source column
        bf16x4 o;
        o[0] = tile[tc4 + 0][c]; o[1] = tile[tc4 + 1][c];
        o[2] = tile[tc4 + 2][c]; o[3] = tile[tc4 + 3][c];
        *(bf16x4*)(d + (size_t)(c0 + c) * R + r0 + tc4) = o;
    }
}

// ---------------------------------------------------------------------------
// GEMM core: 128x128 block tile, 8 wave32, each wave 64x32 (4x2 WMMA accs),
// double-buffered LDS fed by async global->LDS.
// ---------------------------------------------------------------------------
#define GEMM_PROLOG()                                                          \
    const int tid = threadIdx.x, lane = tid & 31, wave = tid >> 5;             \
    const int wm = wave >> 2, wn = wave & 3;                                   \
    const int grp = lane >> 4, kc0 = grp * 8, lr = lane & 15;                  \
    v8f acc[4][2];                                                             \
    _Pragma("unroll") for (int mt = 0; mt < 4; ++mt)                           \
        _Pragma("unroll") for (int nt = 0; nt < 2; ++nt) acc[mt][nt] = (v8f)(0.0f);

#define GEMM_COMPUTE(a, b)                                                     \
    {                                                                          \
        v16bf bfrag[2];                                                        \
        _Pragma("unroll") for (int nt = 0; nt < 2; ++nt)                       \
            bfrag[nt] = load_frag(&(b)[(wn * 32 + nt * 16 + lr) * LDSROW], kc0);\
        _Pragma("unroll") for (int mt = 0; mt < 4; ++mt) {                     \
            v16bf afrag = load_frag(&(a)[(wm * 64 + mt * 16 + lr) * LDSROW], kc0);\
            _Pragma("unroll") for (int nt = 0; nt < 2; ++nt)                   \
                acc[mt][nt] = __builtin_amdgcn_wmma_f32_16x16x32_bf16(         \
                    false, afrag, false, bfrag[nt], (short)0, acc[mt][nt],     \
                    false, false);                                             \
        }                                                                      \
    }

// ---- GEMM1: H = gelu(Xbf @ W1T^T + b1), bf16 out.  grid (DFF/BN, TPE/BM, E)
__global__ __launch_bounds__(256)
void moe_gemm1(const __bf16* __restrict__ Xbf, const __bf16* __restrict__ W1T,
               const float* __restrict__ b1, __bf16* __restrict__ H) {
    __shared__ __align__(16) __bf16 As[2][BM * LDSROW];
    __shared__ __align__(16) __bf16 Bs[2][BN * LDSROW];
    GEMM_PROLOG();
    const int e = blockIdx.z;
    const int rowBase = e * TPE + blockIdx.y * BM;
    const int colBase = blockIdx.x * BN;
    const __bf16* Ab = Xbf + (size_t)rowBase * DMODEL;                 // [m][k]
    const __bf16* Bb = W1T + (size_t)e * DFF * DMODEL
                           + (size_t)colBase * DMODEL;                 // [n][k]
    const int NK = DMODEL / BK;

    stage_tile(Ab, DMODEL, As[0], tid);
    stage_tile(Bb, DMODEL, Bs[0], tid);
    for (int it = 0; it < NK; ++it) {
        wait_async_lds();
        __syncthreads();
        if (it + 1 < NK) {
            stage_tile(Ab + (it + 1) * BK, DMODEL, As[(it + 1) & 1], tid);
            stage_tile(Bb + (it + 1) * BK, DMODEL, Bs[(it + 1) & 1], tid);
        }
        GEMM_COMPUTE(As[it & 1], Bs[it & 1]);
    }

    // epilogue: +bias, gelu, bf16 store.  C/D: VGPR r -> (M=r+8*grp, N=lane%16)
#pragma unroll
    for (int mt = 0; mt < 4; ++mt)
#pragma unroll
        for (int nt = 0; nt < 2; ++nt) {
            const int col  = colBase + wn * 32 + nt * 16 + lr;
            const float bv = b1[e * DFF + col];
#pragma unroll
            for (int r = 0; r < 8; ++r) {
                const int row = rowBase + wm * 64 + mt * 16 + 8 * grp + r;
                H[(size_t)row * DFF + col] = f2bf(gelu_tanh(acc[mt][nt][r] + bv));
            }
        }
}

// ---- GEMM2: OUT = H @ W2T^T + b2, fp32 out.  grid (DMODEL/BN, TPE/BM, E)
__global__ __launch_bounds__(256)
void moe_gemm2(const __bf16* __restrict__ H, const __bf16* __restrict__ W2T,
               const float* __restrict__ b2, float* __restrict__ OUT) {
    __shared__ __align__(16) __bf16 As[2][BM * LDSROW];
    __shared__ __align__(16) __bf16 Bs[2][BN * LDSROW];
    GEMM_PROLOG();
    const int e = blockIdx.z;
    const int rowBase = e * TPE + blockIdx.y * BM;
    const int colBase = blockIdx.x * BN;
    const __bf16* Ab = H   + (size_t)rowBase * DFF;                    // [m][k]
    const __bf16* Bb = W2T + (size_t)e * DMODEL * DFF
                           + (size_t)colBase * DFF;                    // [n][k]
    const int NK = DFF / BK;

    stage_tile(Ab, DFF, As[0], tid);
    stage_tile(Bb, DFF, Bs[0], tid);
    for (int it = 0; it < NK; ++it) {
        wait_async_lds();
        __syncthreads();
        if (it + 1 < NK) {
            stage_tile(Ab + (it + 1) * BK, DFF, As[(it + 1) & 1], tid);
            stage_tile(Bb + (it + 1) * BK, DFF, Bs[(it + 1) & 1], tid);
        }
        GEMM_COMPUTE(As[it & 1], Bs[it & 1]);
    }

#pragma unroll
    for (int mt = 0; mt < 4; ++mt)
#pragma unroll
        for (int nt = 0; nt < 2; ++nt) {
            const int col  = colBase + wn * 32 + nt * 16 + lr;
            const float bv = b2[e * DMODEL + col];
#pragma unroll
            for (int r = 0; r < 8; ++r) {
                const int row = rowBase + wm * 64 + mt * 16 + 8 * grp + r;
                OUT[(size_t)row * DMODEL + col] = acc[mt][nt][r] + bv;
            }
        }
}

// ---------------------------------------------------------------------------
extern "C" void kernel_launch(void* const* d_in, const int* in_sizes, int n_in,
                              void* d_out, int out_size, void* d_ws, size_t ws_size,
                              hipStream_t stream) {
    const float* X  = (const float*)d_in[0];   // [T, D]
    const float* W1 = (const float*)d_in[1];   // [E, D, F]
    const float* b1 = (const float*)d_in[2];   // [E, F]
    const float* W2 = (const float*)d_in[3];   // [E, F, D]
    const float* b2 = (const float*)d_in[4];   // [E, D]
    float* OUT = (float*)d_out;                // [T, D]

    __bf16* Xbf = (__bf16*)d_ws;                                //  64 MB
    __bf16* W1T = Xbf + (size_t)T_TOK * DMODEL;                 // 256 MB [E,F,D]
    __bf16* W2T = W1T + (size_t)NEXP * DMODEL * DFF;            // 256 MB [E,D,F]
    __bf16* H   = W2T + (size_t)NEXP * DFF * DMODEL;            // 256 MB [T,F]

    dim3 blk(256);
    // pass 0: convert / transpose to bf16
    cvt_f32_bf16<<<(unsigned)(((size_t)T_TOK * DMODEL) / 1024), blk, 0, stream>>>(X, Xbf);
    transpose_cvt<<<dim3(DFF / 64, DMODEL / 64, NEXP), blk, 0, stream>>>(W1, W1T, DMODEL, DFF);
    transpose_cvt<<<dim3(DMODEL / 64, DFF / 64, NEXP), blk, 0, stream>>>(W2, W2T, DFF, DMODEL);
    // pass 1/2: WMMA GEMMs
    moe_gemm1<<<dim3(DFF / BN, TPE / BM, NEXP), blk, 0, stream>>>(Xbf, W1T, b1, H);
    moe_gemm2<<<dim3(DMODEL / BN, TPE / BM, NEXP), blk, 0, stream>>>(H, W2T, b2, OUT);
}
